// ConvNeXtMambaForCoCa_76424648065971
// MI455X (gfx1250) — matrix operand order repaired and checked by reading, not compiled, not run on hardware
//
#include <hip/hip_runtime.h>

typedef unsigned short u16;
typedef __bf16 bf16x16 __attribute__((ext_vector_type(16)));
typedef float v8f __attribute__((ext_vector_type(8)));
static_assert(sizeof(bf16x16) == 32, "bf16x16 must be 32 bytes");

// ---------------- device helpers ----------------

__device__ __forceinline__ u16 f2bf(float x) {
  unsigned u = __float_as_uint(x);
  unsigned r = ((u >> 16) & 1u) + 0x7FFFu;
  return (u16)((u + r) >> 16);
}

// act: 0=none 1=gelu(exact) 2=silu 3=softplus 4=sigmoid
__device__ __forceinline__ float act_apply(float x, int act) {
  if (act == 1) return 0.5f * x * (1.0f + erff(x * 0.70710678118654752f));
  if (act == 2) return x * (1.0f / (1.0f + __expf(-x)));
  if (act == 3) return fmaxf(x, 0.0f) + log1pf(__expf(-fabsf(x)));
  if (act == 4) return 1.0f / (1.0f + __expf(-x));
  return x;
}

__device__ __forceinline__ void async_wait0() {
  asm volatile("s_wait_asynccnt 0" ::: "memory");
}

// per-lane async memory->LDS 16B copy (ASYNCcnt-tracked, no VGPR round trip)
__device__ __forceinline__ void async_b128(void* lds_dst, const void* gsrc) {
  unsigned ldsoff = (unsigned)(unsigned long long)lds_dst;   // low 32 bits = LDS offset
  unsigned long long ga = (unsigned long long)gsrc;
  asm volatile("global_load_async_to_lds_b128 %0, %1, off"
               :: "v"(ldsoff), "v"(ga) : "memory");
}

// ---------------- WMMA GEMM: out = act(A[M,K](bf16) * Bw[N,K](bf16)^T + bias) (+resid) ----------------
// Tile 128x128, K-step 32. 256 threads = 8 waves arranged 4(M) x 2(N); each wave 32x64 = 2x4 WMMA tiles.
// Double-buffered LDS staging via GLOBAL_LOAD_ASYNC_TO_LDS_B128; scalar zero-pad path for edge tiles.

__global__ void __launch_bounds__(256)
k_gemm(const u16* __restrict__ A, int lda,
       const u16* __restrict__ Bw, int ldb,
       float* __restrict__ Cf, u16* __restrict__ Ch, int ldc,
       const float* __restrict__ bias, const float* __restrict__ resid,
       int M, int N, int K, int act)
{
  __shared__ __align__(16) u16 As[2][128][32];
  __shared__ __align__(16) u16 Bs[2][128][32];

  const int tid  = threadIdx.x;
  const int lane = tid & 31;
  const int wave = tid >> 5;
  const int wr   = wave >> 1;     // 0..3 : wave row (M)
  const int wc   = wave & 1;      // 0..1 : wave col (N)
  const int lh   = lane >> 4;     // lane half 0/1
  const int lm   = lane & 15;
  const long long tileM = (long long)blockIdx.y * 128;
  const long long tileN = (long long)blockIdx.x * 128;

  const bool avec = ((lda & 7) == 0);
  const bool bvec = ((ldb & 7) == 0);

  // stage one 128x32 K-tile of A and B into LDS buffer `buf`
  auto stage = [&](int buf, int k0) {
    for (int i = tid; i < 512; i += 256) {
      int r = i >> 2, c8 = (i & 3) << 3;
      long long gr = tileM + r;
      int gc = k0 + c8;
      u16* dst = &As[buf][r][c8];
      if (gr < M && avec && gc + 8 <= K) {
        async_b128(dst, A + gr * (long long)lda + gc);
      } else {
        union { u16 u[8]; uint4 q; } tmp;
        for (int j = 0; j < 8; ++j)
          tmp.u[j] = (gr < M && gc + j < K) ? A[gr * (long long)lda + gc + j] : (u16)0;
        *(uint4*)dst = tmp.q;
      }
    }
    for (int i = tid; i < 512; i += 256) {
      int r = i >> 2, c8 = (i & 3) << 3;
      long long gr = tileN + r;
      int gc = k0 + c8;
      u16* dst = &Bs[buf][r][c8];
      if (gr < N && bvec && gc + 8 <= K) {
        async_b128(dst, Bw + gr * (long long)ldb + gc);
      } else {
        union { u16 u[8]; uint4 q; } tmp;
        for (int j = 0; j < 8; ++j)
          tmp.u[j] = (gr < N && gc + j < K) ? Bw[gr * (long long)ldb + gc + j] : (u16)0;
        *(uint4*)dst = tmp.q;
      }
    }
  };

  v8f acc[2][4];
  for (int mi = 0; mi < 2; ++mi)
    for (int ni = 0; ni < 4; ++ni)
      for (int v = 0; v < 8; ++v) acc[mi][ni][v] = 0.0f;

  stage(0, 0);
  async_wait0();
  __syncthreads();

  int p = 0;
  for (int k0 = 0; k0 < K; k0 += 32) {
    // prefetch next K-tile into the other buffer while this one is consumed
    if (k0 + 32 < K) stage(1 - p, k0 + 32);

    // A fragment (16x32): lane -> m = lm, element j -> k = (j&7) + lh*8 + (j>>3)*16
    union { uint4 q[2]; bf16x16 v; } afr[2], bfr[4];
    for (int mi = 0; mi < 2; ++mi) {
      const u16* fp = &As[p][wr * 32 + mi * 16 + lm][lh * 8];
      afr[mi].q[0] = *(const uint4*)fp;
      afr[mi].q[1] = *(const uint4*)(fp + 16);
    }
    // B fragment (32x16): lane -> n = lm, element j -> k = (j&7) + lh*8 + (j>>3)*16
    for (int ni = 0; ni < 4; ++ni) {
      const u16* fp = &Bs[p][wc * 64 + ni * 16 + lm][lh * 8];
      bfr[ni].q[0] = *(const uint4*)fp;
      bfr[ni].q[1] = *(const uint4*)(fp + 16);
    }
    for (int mi = 0; mi < 2; ++mi)
      for (int ni = 0; ni < 4; ++ni)
        acc[mi][ni] = __builtin_amdgcn_wmma_f32_16x16x32_bf16(
            false, afr[mi].v, false, bfr[ni].v, (short)0, acc[mi][ni], false, false);

    // next buffer's async fills must have landed, and all waves be done reading buf p
    async_wait0();
    __syncthreads();
    p ^= 1;
  }

  // epilogue: C/D layout: vgpr v -> row (lh*8 + v), col = lm
  for (int ni = 0; ni < 4; ++ni) {
    long long n = tileN + wc * 64 + ni * 16 + lm;
    if (n >= N) continue;
    float bv = bias ? bias[n] : 0.0f;
    for (int mi = 0; mi < 2; ++mi) {
      for (int v = 0; v < 8; ++v) {
        long long m = tileM + wr * 32 + mi * 16 + lh * 8 + v;
        if (m >= M) continue;
        float x = acc[(p ^ (K > 0)) ? mi : mi][ni][v];  // p unused; keep indexing simple
        x = acc[mi][ni][v] + bv;
        x = act_apply(x, act);
        if (resid) x += resid[m * (long long)ldc + n];
        if (Cf) Cf[m * (long long)ldc + n] = x;
        if (Ch) Ch[m * (long long)ldc + n] = f2bf(x);
      }
    }
  }
}

// ---------------- elementwise / conv / norm kernels ----------------

__global__ void k_cvt(const float* __restrict__ x, u16* __restrict__ y, long long n) {
  long long i = (long long)blockIdx.x * blockDim.x + threadIdx.x;
  if (i < n) y[i] = f2bf(x[i]);
}

__global__ void k_aneg(const float* __restrict__ x, float* __restrict__ y, long long n) {
  long long i = (long long)blockIdx.x * blockDim.x + threadIdx.x;
  if (i < n) y[i] = -expf(x[i]);
}

// stem: x NCHW (8,3,224,224) f32 -> y NHWC (8,56,56,128)
__global__ void k_stem(const float* __restrict__ x, const float* __restrict__ wt,
                       const float* __restrict__ bs, float* __restrict__ y, long long total) {
  long long i = (long long)blockIdx.x * blockDim.x + threadIdx.x;
  if (i >= total) return;
  int co = (int)(i % 128); long long t = i / 128;
  int wo = (int)(t % 56); t /= 56;
  int ho = (int)(t % 56); int b = (int)(t / 56);
  float acc = bs[co];
  for (int ci = 0; ci < 3; ++ci)
    for (int fh = 0; fh < 4; ++fh)
      for (int fw = 0; fw < 4; ++fw)
        acc += wt[((co * 3 + ci) * 4 + fh) * 4 + fw] *
               x[(((long long)b * 3 + ci) * 224 + (ho * 4 + fh)) * 224 + (wo * 4 + fw)];
  y[i] = acc;
}

// depthwise 7x7, NHWC, pad 3
__global__ void k_dw(const float* __restrict__ x, const float* __restrict__ wt,
                     const float* __restrict__ bs, float* __restrict__ y,
                     int B, int H, int Wd, int C, long long total) {
  long long i = (long long)blockIdx.x * blockDim.x + threadIdx.x;
  if (i >= total) return;
  int c = (int)(i % C); long long t = i / C;
  int wx = (int)(t % Wd); t /= Wd;
  int hx = (int)(t % H); int b = (int)(t / H);
  float acc = bs[c];
  for (int fh = 0; fh < 7; ++fh) {
    int h2 = hx + fh - 3;
    if ((unsigned)h2 >= (unsigned)H) continue;
    for (int fw = 0; fw < 7; ++fw) {
      int w2 = wx + fw - 3;
      if ((unsigned)w2 >= (unsigned)Wd) continue;
      acc += wt[c * 49 + fh * 7 + fw] * x[(((long long)b * H + h2) * Wd + w2) * C + c];
    }
  }
  y[i] = acc;
}

// dense 2x2 stride-2 downsample conv, NHWC
__global__ void k_down(const float* __restrict__ x, const float* __restrict__ wt,
                       const float* __restrict__ bs, float* __restrict__ y,
                       int B, int H, int Wd, int Ci, int Co, long long total) {
  long long i = (long long)blockIdx.x * blockDim.x + threadIdx.x;
  if (i >= total) return;
  int Ho = H >> 1, Wo = Wd >> 1;
  int co = (int)(i % Co); long long t = i / Co;
  int wo = (int)(t % Wo); t /= Wo;
  int ho = (int)(t % Ho); int b = (int)(t / Ho);
  float acc = bs[co];
  const float* xp0 = x + (((long long)b * H + ho * 2) * Wd + wo * 2) * Ci;
  long long rs = (long long)Wd * Ci;
  for (int ci = 0; ci < Ci; ++ci) {
    const float* wp = wt + ((long long)co * Ci + ci) * 4;
    acc += wp[0] * xp0[ci] + wp[1] * xp0[Ci + ci] + wp[2] * xp0[rs + ci] + wp[3] * xp0[rs + Ci + ci];
  }
  y[i] = acc;
}

// wave-per-row LayerNorm over last dim C (C multiple of 32); optional f32 and/or bf16 output
__global__ void k_ln(const float* __restrict__ x, const float* __restrict__ g,
                     const float* __restrict__ bt, float* __restrict__ yf,
                     u16* __restrict__ yh, int rows, int C) {
  long long gt = (long long)blockIdx.x * blockDim.x + threadIdx.x;
  int row = (int)(gt >> 5);
  int lane = (int)(gt & 31);
  if (row >= rows) return;
  const float* r = x + (long long)row * C;
  float s = 0.0f;
  for (int c = lane; c < C; c += 32) s += r[c];
  for (int m = 16; m >= 1; m >>= 1) s += __shfl_xor(s, m, 32);
  float mean = s / C;
  float v = 0.0f;
  for (int c = lane; c < C; c += 32) { float d = r[c] - mean; v += d * d; }
  for (int m = 16; m >= 1; m >>= 1) v += __shfl_xor(v, m, 32);
  float inv = rsqrtf(v / C + 1e-6f);
  for (int c = lane; c < C; c += 32) {
    float o = (r[c] - mean) * inv * g[c] + bt[c];
    long long idx = (long long)row * C + c;
    if (yf) yf[idx] = o;
    if (yh) yh[idx] = f2bf(o);
  }
}

// causal depthwise conv1d (K=4, left pad 3) over xin = xz[:, :di], fused SiLU; writes f32 + bf16
__global__ void k_conv1d(const float* __restrict__ xz, const float* __restrict__ cw,
                         const float* __restrict__ cb, float* __restrict__ uf,
                         u16* __restrict__ ub, int B, int L, int di, long long total) {
  long long i = (long long)blockIdx.x * blockDim.x + threadIdx.x;
  if (i >= total) return;
  int d = (int)(i % di); long long t = i / di;
  int l = (int)(t % L); int b = (int)(t / L);
  int twodi = 2 * di;
  const float* bp = xz + ((long long)b * L) * twodi + d;
  float acc = cb[d];
#pragma unroll
  for (int k = 0; k < 4; ++k) {
    int l2 = l - 3 + k;
    if (l2 >= 0) acc += cw[d * 4 + k] * bp[(long long)l2 * twodi];
  }
  float s = acc * (1.0f / (1.0f + __expf(-acc)));
  uf[i] = s;
  ub[i] = f2bf(s);
}

// selective scan: one thread per (b, d); 16-wide state in VGPRs; fuses + u*D.
// global_prefetch the next timestep's operand rows to hide latency behind the exp/fma chain.
__global__ void k_scan(const float* __restrict__ dt, const float* __restrict__ u,
                       const float* __restrict__ dbl, int ldd, int dtr,
                       const float* __restrict__ Aneg, const float* __restrict__ Dp,
                       float* __restrict__ y, int B, int L, int di) {
  long long i = (long long)blockIdx.x * blockDim.x + threadIdx.x;
  if (i >= (long long)B * di) return;
  int d = (int)(i % di); int b = (int)(i / di);
  float A[16], h[16];
#pragma unroll
  for (int s = 0; s < 16; ++s) { A[s] = Aneg[d * 16 + s]; h[s] = 0.0f; }
  float Dv = Dp[d];
  for (int l = 0; l < L; ++l) {
    long long n = (long long)b * L + l;
    if (l + 1 < L) {
      __builtin_prefetch(dt + (n + 1) * di + d, 0, 1);
      __builtin_prefetch(u + (n + 1) * di + d, 0, 1);
      __builtin_prefetch(dbl + (n + 1) * (long long)ldd + dtr, 0, 1);
    }
    float dtv = dt[n * di + d];
    float uv = u[n * di + d];
    const float* bc = dbl + n * (long long)ldd + dtr;
    float du = dtv * uv;
    float acc = 0.0f;
#pragma unroll
    for (int s = 0; s < 16; ++s) {
      float hh = __expf(dtv * A[s]) * h[s] + du * bc[s];
      h[s] = hh;
      acc += hh * bc[16 + s];
    }
    y[n * di + d] = acc + uv * Dv;
  }
}

// gate: yg = bf16( yD * silu(z) ), z = xz[:, di + d]
__global__ void k_gate(const float* __restrict__ yD, const float* __restrict__ xz,
                       u16* __restrict__ yg, int di, long long total) {
  long long i = (long long)blockIdx.x * blockDim.x + threadIdx.x;
  if (i >= total) return;
  int d = (int)(i % di); long long n = i / di;
  float z = xz[n * (2LL * di) + di + d];
  float g = z * (1.0f / (1.0f + __expf(-z)));
  yg[i] = f2bf(yD[i] * g);
}

// mean over HW: pooled[b,o]
__global__ void k_pool(const float* __restrict__ t, float* __restrict__ o,
                       int HW, int C, long long total) {
  long long i = (long long)blockIdx.x * blockDim.x + threadIdx.x;
  if (i >= total) return;
  int c = (int)(i % C); int b = (int)(i / C);
  float s = 0.0f;
  for (int hw = 0; hw < HW; ++hw) s += t[((long long)b * HW + hw) * C + c];
  o[i] = s / (float)HW;
}

__global__ void k_scale(float* __restrict__ t, const float* __restrict__ a,
                        int HW, int C, long long total) {
  long long i = (long long)blockIdx.x * blockDim.x + threadIdx.x;
  if (i >= total) return;
  int c = (int)(i % C);
  int b = (int)(i / ((long long)HW * C));
  t[i] *= a[(long long)b * C + c];
}

// bilinear (half-pixel, edge clamp) resize of src (B,Hi,Wi,C) added into dst (B,Ho,Wo,C)
__global__ void k_resize_add(const float* __restrict__ src, float* __restrict__ dst,
                             int B, int Hi, int Wi, int Ho, int Wo, int C, long long total) {
  long long i = (long long)blockIdx.x * blockDim.x + threadIdx.x;
  if (i >= total) return;
  int c = (int)(i % C); long long t = i / C;
  int wo = (int)(t % Wo); t /= Wo;
  int ho = (int)(t % Ho); int b = (int)(t / Ho);
  float sy = (ho + 0.5f) * ((float)Hi / (float)Ho) - 0.5f;
  float sx = (wo + 0.5f) * ((float)Wi / (float)Wo) - 0.5f;
  sy = fminf(fmaxf(sy, 0.0f), (float)(Hi - 1));
  sx = fminf(fmaxf(sx, 0.0f), (float)(Wi - 1));
  int y0 = (int)sy, x0 = (int)sx;
  int y1 = min(y0 + 1, Hi - 1), x1 = min(x0 + 1, Wi - 1);
  float fy = sy - y0, fx = sx - x0;
  const float* s = src + ((long long)b * Hi) * Wi * C + c;
  float v00 = s[((long long)y0 * Wi + x0) * C];
  float v01 = s[((long long)y0 * Wi + x1) * C];
  float v10 = s[((long long)y1 * Wi + x0) * C];
  float v11 = s[((long long)y1 * Wi + x1) * C];
  float top = v00 + fx * (v01 - v00);
  float bot = v10 + fx * (v11 - v10);
  dst[i] += top + fy * (bot - top);
}

// ---------------- host orchestration ----------------

extern "C" void kernel_launch(void* const* d_in, const int* in_sizes, int n_in,
                              void* d_out, int out_size, void* d_ws, size_t ws_size,
                              hipStream_t stream) {
  (void)in_sizes; (void)n_in; (void)out_size; (void)ws_size;
  const int CH[4] = {128, 256, 512, 512};
  const int HH[4] = {56, 28, 14, 7};
  const int DEP[4] = {3, 3, 9, 3};
  const int Bsz = 8;

  // ---- unpack flattened inputs (dict order) ----
  int ii = 0;
  auto inF = [&]() -> const float* { return (const float*)d_in[ii++]; };
  const float* X = inF();
  const float* stem_w = inF(); const float* stem_b = inF();
  const float* stem_lw = inF(); const float* stem_lb = inF();
  const float *dn_lw[3], *dn_lb[3], *dn_w[3], *dn_b[3];
  for (int i = 0; i < 3; ++i) { dn_lw[i] = inF(); dn_lb[i] = inF(); dn_w[i] = inF(); dn_b[i] = inF(); }
  enum { DW_W, DW_B, N_W, N_B, P1_W, P1_B, P2_W, P2_B, IN_W, CV_W, CV_B,
         XP_W, DT_W, DT_B, A_LOG, D_P, OUT_W, NPAR };
  const float* blk[18][NPAR];
  {
    int nb = 0;
    for (int s = 0; s < 4; ++s)
      for (int d = 0; d < DEP[s]; ++d, ++nb)
        for (int t = 0; t < NPAR; ++t) blk[nb][t] = inF();
  }
  const float *lt_w[4], *lt_b[4], *lt_aw[4], *lt_ab[4];
  for (int i = 0; i < 4; ++i) { lt_w[i] = inF(); lt_b[i] = inF(); lt_aw[i] = inF(); lt_ab[i] = inF(); }
  const float* fin_w = inF(); const float* fin_b = inF();

  // ---- workspace bump allocator ----
  char* base = (char*)d_ws;
  size_t off = 0;
  auto alloc = [&](size_t bytes) -> void* {
    off = (off + 255) & ~(size_t)255;
    void* p = base + off;
    off += bytes;
    return p;
  };
  auto allocF = [&](size_t n) -> float* { return (float*)alloc(n * 4); };
  auto allocH = [&](size_t n) -> u16* { return (u16*)alloc(n * 2); };

  auto grid1 = [](long long n) { return dim3((unsigned)((n + 255) / 256)); };
  auto cvt = [&](const float* s, u16* d, long long n) {
    k_cvt<<<grid1(n), 256, 0, stream>>>(s, d, n);
  };
  auto gemm = [&](const u16* A, int lda, const u16* Bw, int ldb,
                  float* Cf, u16* Ch, int ldc, const float* bias, const float* resid,
                  long long M, int N, int K, int act) {
    dim3 g((unsigned)((N + 127) / 128), (unsigned)((M + 127) / 128));
    k_gemm<<<g, 256, 0, stream>>>(A, lda, Bw, ldb, Cf, Ch, ldc, bias, resid, (int)M, N, K, act);
  };
  auto lnrow = [&](const float* x, const float* g, const float* b,
                   float* yf, u16* yh, long long rows, int C) {
    k_ln<<<grid1(rows * 32), 256, 0, stream>>>(x, g, b, yf, yh, (int)rows, C);
  };

  // ---- persistent: bf16 weights + A = -exp(A_log) + stage ping-pong buffers ----
  u16 *Wp1[18], *Wp2[18], *Win[18], *Wxp[18], *Wdt[18], *Wout[18];
  float* Anegs[18];
  {
    int nb = 0;
    for (int s = 0; s < 4; ++s) {
      int C = CH[s], di = 2 * C, dtr = C / 16, nd = dtr + 32;
      for (int d = 0; d < DEP[s]; ++d, ++nb) {
        Wp1[nb] = allocH((size_t)4 * C * C);  cvt(blk[nb][P1_W], Wp1[nb], (long long)4 * C * C);
        Wp2[nb] = allocH((size_t)4 * C * C);  cvt(blk[nb][P2_W], Wp2[nb], (long long)4 * C * C);
        Win[nb] = allocH((size_t)2 * di * C); cvt(blk[nb][IN_W], Win[nb], (long long)2 * di * C);
        Wxp[nb] = allocH((size_t)nd * di);    cvt(blk[nb][XP_W], Wxp[nb], (long long)nd * di);
        Wdt[nb] = allocH((size_t)di * dtr);   cvt(blk[nb][DT_W], Wdt[nb], (long long)di * dtr);
        Wout[nb] = allocH((size_t)C * di);    cvt(blk[nb][OUT_W], Wout[nb], (long long)C * di);
        Anegs[nb] = allocF((size_t)di * 16);
        k_aneg<<<grid1((long long)di * 16), 256, 0, stream>>>(blk[nb][A_LOG], Anegs[nb], (long long)di * 16);
      }
    }
  }
  u16 *Lw[4], *Law[4];
  for (int i = 0; i < 4; ++i) {
    Lw[i] = allocH((size_t)512 * CH[i]); cvt(lt_w[i], Lw[i], (long long)512 * CH[i]);
    Law[i] = allocH((size_t)512 * 512);  cvt(lt_aw[i], Law[i], 512 * 512);
  }
  float *xa[4], *xb[4];
  for (int s = 0; s < 4; ++s) {
    size_t n = (size_t)Bsz * HH[s] * HH[s] * CH[s];
    xa[s] = allocF(n); xb[s] = allocF(n);
  }

  // ---- stages ----
  float* feat[4];
  for (int s = 0; s < 4; ++s) {
    int C = CH[s], H = HH[s], Wd = HH[s];
    long long L = (long long)H * Wd;
    long long Nt = (long long)Bsz * L;
    int di = 2 * C, dtr = C / 16, nd = dtr + 32;
    float *cur, *nxt;
    if (s == 0) {
      k_stem<<<grid1(Nt * C), 256, 0, stream>>>(X, stem_w, stem_b, xb[0], Nt * C);
      lnrow(xb[0], stem_lw, stem_lb, xa[0], nullptr, Nt, C);
      cur = xa[0]; nxt = xb[0];
    } else {
      int Cp = CH[s - 1], Hp = HH[s - 1];
      long long Ntp = (long long)Bsz * Hp * Hp;
      float* scr = (feat[s - 1] == xa[s - 1]) ? xb[s - 1] : xa[s - 1];
      lnrow(feat[s - 1], dn_lw[s - 1], dn_lb[s - 1], scr, nullptr, Ntp, Cp);
      k_down<<<grid1(Nt * C), 256, 0, stream>>>(scr, dn_w[s - 1], dn_b[s - 1], xa[s],
                                                Bsz, Hp, Hp, Cp, C, Nt * C);
      cur = xa[s]; nxt = xb[s];
    }
    int b0 = 0; for (int t = 0; t < s; ++t) b0 += DEP[t];
    for (int d = 0; d < DEP[s]; ++d) {
      int bi = b0 + d;
      const float* const* P = blk[bi];
      size_t mark = off;

      float* dwo = allocF((size_t)Nt * C);
      k_dw<<<grid1(Nt * C), 256, 0, stream>>>(cur, P[DW_W], P[DW_B], dwo, Bsz, H, Wd, C, Nt * C);
      u16* lnb = allocH((size_t)Nt * C);
      lnrow(dwo, P[N_W], P[N_B], nullptr, lnb, Nt, C);
      u16* p1o = allocH((size_t)Nt * 4 * C);
      gemm(lnb, C, Wp1[bi], C, nullptr, p1o, 4 * C, P[P1_B], nullptr, Nt, 4 * C, C, 1);
      u16* p2o = allocH((size_t)Nt * C);
      gemm(p1o, 4 * C, Wp2[bi], 4 * C, nullptr, p2o, C, P[P2_B], nullptr, Nt, C, 4 * C, 0);
      float* xz = allocF((size_t)Nt * 2 * di);
      gemm(p2o, C, Win[bi], C, xz, nullptr, 2 * di, nullptr, nullptr, Nt, 2 * di, C, 0);
      float* uf = allocF((size_t)Nt * di);
      u16* ub = allocH((size_t)Nt * di);
      k_conv1d<<<grid1(Nt * di), 256, 0, stream>>>(xz, P[CV_W], P[CV_B], uf, ub,
                                                   Bsz, (int)L, di, Nt * di);
      float* dbl = allocF((size_t)Nt * nd);
      gemm(ub, di, Wxp[bi], di, dbl, nullptr, nd, nullptr, nullptr, Nt, nd, di, 0);
      u16* dblh = allocH((size_t)Nt * nd);
      cvt(dbl, dblh, Nt * nd);
      float* dtb = allocF((size_t)Nt * di);
      gemm(dblh, nd, Wdt[bi], dtr, dtb, nullptr, di, P[DT_B], nullptr, Nt, di, dtr, 3);
      float* yD = allocF((size_t)Nt * di);
      k_scan<<<grid1((long long)Bsz * di), 256, 0, stream>>>(dtb, uf, dbl, nd, dtr,
                                                             Anegs[bi], P[D_P], yD, Bsz, (int)L, di);
      u16* yg = allocH((size_t)Nt * di);
      k_gate<<<grid1(Nt * di), 256, 0, stream>>>(yD, xz, yg, di, Nt * di);
      gemm(yg, di, Wout[bi], di, nxt, nullptr, C, nullptr, cur /*residual*/, Nt, C, di, 0);

      off = mark;
      float* tp = cur; cur = nxt; nxt = tp;
    }
    feat[s] = cur;
  }

  // ---- aggregate (FPN-style) ----
  float* t[4];
  for (int i = 0; i < 4; ++i) t[i] = allocF((size_t)Bsz * HH[i] * HH[i] * 512);
  float* pooled = allocF((size_t)Bsz * 512);
  u16* pooledh = allocH((size_t)Bsz * 512);
  float* att = allocF((size_t)Bsz * 512);
  for (int i = 0; i < 4; ++i) {
    int Ci = CH[i];
    long long HW = (long long)HH[i] * HH[i];
    long long Nt = (long long)Bsz * HW;
    size_t m2 = off;
    u16* fh = allocH((size_t)Nt * Ci);
    cvt(feat[i], fh, Nt * Ci);
    gemm(fh, Ci, Lw[i], Ci, t[i], nullptr, 512, lt_b[i], nullptr, Nt, 512, Ci, 0);
    k_pool<<<grid1((long long)Bsz * 512), 256, 0, stream>>>(t[i], pooled, (int)HW, 512,
                                                            (long long)Bsz * 512);
    cvt(pooled, pooledh, (long long)Bsz * 512);
    gemm(pooledh, 512, Law[i], 512, att, nullptr, 512, lt_ab[i], nullptr, Bsz, 512, 512, 4);
    k_scale<<<grid1(Nt * 512), 256, 0, stream>>>(t[i], att, (int)HW, 512, Nt * 512);
    off = m2;
  }
  for (int i = 3; i >= 1; --i) {
    long long tot = (long long)Bsz * HH[i - 1] * HH[i - 1] * 512;
    k_resize_add<<<grid1(tot), 256, 0, stream>>>(t[i], t[i - 1], Bsz, HH[i], HH[i],
                                                 HH[i - 1], HH[i - 1], 512, tot);
  }

  // final LN -> d_out (B, 3136, 512) f32
  lnrow(t[0], fin_w, fin_b, (float*)d_out, nullptr, (long long)Bsz * HH[0] * HH[0], 512);
}
